// GATLayer_22119081575271
// MI455X (gfx1250) — matrix operand order repaired and checked by reading, not compiled, not run on hardware
//
#include <hip/hip_runtime.h>
#include <hip/hip_bf16.h>

typedef __bf16 v16bf __attribute__((ext_vector_type(16)));
typedef __bf16 v8bf  __attribute__((ext_vector_type(8)));
typedef float  v8f   __attribute__((ext_vector_type(8)));

#define NEG_GATV2 0.2f
#define NEG_POST  0.01f

// ---------------------------------------------------------------------------
// Kernel 1: prep — convert x and weights to bf16, init emax/denom/out.
// ---------------------------------------------------------------------------
__global__ void gat_prep_kernel(const float* __restrict__ x,
                                const float* __restrict__ Wl,
                                const float* __restrict__ Wr,
                                const float* __restrict__ Wsk,
                                __hip_bfloat16* __restrict__ xbf,
                                __hip_bfloat16* __restrict__ wcat,
                                float* __restrict__ emax,
                                float* __restrict__ denom,
                                float* __restrict__ out,
                                int nx, int nh)
{
    int i = blockIdx.x * blockDim.x + threadIdx.x;
    if (i < nx) {
        xbf[i] = __float2bfloat16(x[i]);
        out[i] = 0.0f;
    }
    if (i < 128 * 128) {
        wcat[i]             = __float2bfloat16(Wl[i]);
        wcat[i + 128 * 128] = __float2bfloat16(Wr[i]);
        wcat[i + 256 * 128] = __float2bfloat16(Wsk[i]);
    }
    if (i < nh) {
        emax[i]  = __uint_as_float(0xFF800000u);  // -inf
        denom[i] = 0.0f;
    }
}

// ---------------------------------------------------------------------------
// Kernel 2: fused WMMA GEMM: [xl | xr | xs] = x @ [W_l | W_r | W_skip].T
// Block = 256 threads (8 waves), each block owns 16 rows, each wave covers
// 3 of the 24 column tiles. bf16 inputs, f32 accumulate.
// ---------------------------------------------------------------------------
__global__ __launch_bounds__(256) void gat_gemm_wmma_kernel(
    const __bf16* __restrict__ xbf,     // [N,128] bf16
    const __bf16* __restrict__ wcat,    // [384,128] bf16 (row o = output col)
    const float*  __restrict__ b_l,     // [128]
    const float*  __restrict__ b_r,     // [128]
    float* __restrict__ xl,             // [N,128]
    float* __restrict__ xr,             // [N,128]
    float* __restrict__ xs)             // [N,128]
{
    const int row0 = blockIdx.x * 16;
    const int wave = threadIdx.x >> 5;
    const int lane = threadIdx.x & 31;
    const int m    = lane & 15;   // A row / B col / D col within tile
    const int kh   = lane >> 4;   // K-half select

    // --- Preload the 4 A fragments covering K=0..127 (contiguous layout:
    // elems 0..15 <- K[16*kh, 16*kh+16)) ---
    v16bf afr[4];
    const __bf16* arow = xbf + (size_t)(row0 + m) * 128;
#pragma unroll
    for (int kb = 0; kb < 4; ++kb) {
        const __bf16* p = arow + kb * 32 + 16 * kh;
        v8bf lo = *(const v8bf*)(p);
        v8bf hi = *(const v8bf*)(p + 8);
#pragma unroll
        for (int i = 0; i < 8; ++i) {
            afr[kb][i]     = lo[i];
            afr[kb][i + 8] = hi[i];
        }
    }

#pragma unroll
    for (int t = 0; t < 3; ++t) {
        const int ct   = wave + 8 * t;      // column tile 0..23 (wave-uniform)
        const int col0 = ct * 16;
        const int o    = col0 + m;          // global output column 0..383

        // B fragment: B[k,n] = wcat[o, k]; per lane one contiguous 16-elem
        // chunk at K offset 16*kh (contiguous K order, same as A: lanes 0-15 K=0..15, 16-31 K=16..31)
        const __bf16* brow = wcat + (size_t)o * 128 + 16 * kh;

        v8f acc = {0.f, 0.f, 0.f, 0.f, 0.f, 0.f, 0.f, 0.f};
#pragma unroll
        for (int kb = 0; kb < 4; ++kb) {
            v16bf bfr = *(const v16bf*)(brow + kb * 32);
            acc = __builtin_amdgcn_wmma_f32_16x16x32_bf16(
                false, afr[kb], false, bfr, (short)0, acc, false, false);
        }

        // Which output buffer (wave-uniform: tiles never straddle 128-boundaries)
        float  bv;
        float* outp;
        int    col;
        if (o < 128)      { bv = b_l[o];       outp = xl; col = o;       }
        else if (o < 256) { bv = b_r[o - 128]; outp = xr; col = o - 128; }
        else              { bv = 0.0f;         outp = xs; col = o - 256; }

        // D layout: acc[v] -> row (row0 + v + 8*kh), col (col0 + m)
#pragma unroll
        for (int v = 0; v < 8; ++v) {
            int rr = row0 + v + 8 * kh;
            outp[(size_t)rr * 128 + col] = acc[v] + bv;
        }
    }
}

// ---------------------------------------------------------------------------
// float atomic max via signed/unsigned int trick
// ---------------------------------------------------------------------------
__device__ __forceinline__ void atomicMaxF(float* addr, float val)
{
    if (val >= 0.0f) atomicMax((int*)addr, __float_as_int(val));
    else             atomicMin((unsigned int*)addr, __float_as_uint(val));
}

// ---------------------------------------------------------------------------
// Kernel 3: per-(edge,head) attention score e = att . leaky(xl[src]+xr[dst])
// plus segment max via atomics. Self loops are indices [E, E+N).
// ---------------------------------------------------------------------------
__global__ void gat_edge_score_kernel(const int* __restrict__ srcp,
                                      const int* __restrict__ dstp,
                                      const float* __restrict__ xl,
                                      const float* __restrict__ xr,
                                      const float* __restrict__ att,
                                      float* __restrict__ ebuf,
                                      float* __restrict__ emax,
                                      int E, int EN)
{
    int idx = blockIdx.x * blockDim.x + threadIdx.x;
    if (idx >= EN * 4) return;
    int eidx = idx >> 2;
    int h    = idx & 3;
    int src  = (eidx < E) ? srcp[eidx] : (eidx - E);
    int dst  = (eidx < E) ? dstp[eidx] : (eidx - E);

    const float4* pl = (const float4*)(xl + (size_t)src * 128 + h * 32);
    const float4* pr = (const float4*)(xr + (size_t)dst * 128 + h * 32);
    const float4* pa = (const float4*)(att + h * 32);

    float s = 0.0f;
#pragma unroll
    for (int j = 0; j < 8; ++j) {
        float4 l = pl[j], r = pr[j], a = pa[j];
        float v0 = l.x + r.x; v0 = v0 > 0.f ? v0 : NEG_GATV2 * v0;
        float v1 = l.y + r.y; v1 = v1 > 0.f ? v1 : NEG_GATV2 * v1;
        float v2 = l.z + r.z; v2 = v2 > 0.f ? v2 : NEG_GATV2 * v2;
        float v3 = l.w + r.w; v3 = v3 > 0.f ? v3 : NEG_GATV2 * v3;
        s += a.x * v0 + a.y * v1 + a.z * v2 + a.w * v3;
    }
    ebuf[idx] = s;
    atomicMaxF(&emax[dst * 4 + h], s);
}

// ---------------------------------------------------------------------------
// Kernel 4: ex = exp(e - emax[dst]); segment-sum into denom
// ---------------------------------------------------------------------------
__global__ void gat_edge_exp_kernel(const int* __restrict__ dstp,
                                    float* __restrict__ ebuf,
                                    const float* __restrict__ emax,
                                    float* __restrict__ denom,
                                    int E, int EN)
{
    int idx = blockIdx.x * blockDim.x + threadIdx.x;
    if (idx >= EN * 4) return;
    int eidx = idx >> 2;
    int h    = idx & 3;
    int dst  = (eidx < E) ? dstp[eidx] : (eidx - E);

    float ex = __expf(ebuf[idx] - emax[dst * 4 + h]);
    ebuf[idx] = ex;
    atomicAdd(&denom[dst * 4 + h], ex);
}

// ---------------------------------------------------------------------------
// Kernel 5: alpha = ex/denom; scatter-add alpha * xl[src] into out[dst]
// ---------------------------------------------------------------------------
__global__ void gat_edge_agg_kernel(const int* __restrict__ srcp,
                                    const int* __restrict__ dstp,
                                    const float* __restrict__ xl,
                                    const float* __restrict__ ebuf,
                                    const float* __restrict__ denom,
                                    float* __restrict__ out,
                                    int E, int EN)
{
    int idx = blockIdx.x * blockDim.x + threadIdx.x;
    if (idx >= EN * 4) return;
    int eidx = idx >> 2;
    int h    = idx & 3;
    int src  = (eidx < E) ? srcp[eidx] : (eidx - E);
    int dst  = (eidx < E) ? dstp[eidx] : (eidx - E);

    float alpha = ebuf[idx] / (denom[dst * 4 + h] + 1e-16f);
    const float4* pl = (const float4*)(xl + (size_t)src * 128 + h * 32);
    float* po = out + (size_t)dst * 128 + h * 32;
#pragma unroll
    for (int j = 0; j < 8; ++j) {
        float4 lv = pl[j];
        atomicAdd(po + 4 * j + 0, alpha * lv.x);
        atomicAdd(po + 4 * j + 1, alpha * lv.y);
        atomicAdd(po + 4 * j + 2, alpha * lv.z);
        atomicAdd(po + 4 * j + 3, alpha * lv.w);
    }
}

// ---------------------------------------------------------------------------
// Kernel 6: out = leaky(out + bias, 0.01) + x@W_skip.T  (in place on d_out)
// ---------------------------------------------------------------------------
__global__ void gat_final_kernel(float* __restrict__ out,
                                 const float* __restrict__ bias,
                                 const float* __restrict__ xs,
                                 int nx)
{
    int i = blockIdx.x * blockDim.x + threadIdx.x;
    if (i >= nx) return;
    float v = out[i] + bias[i & 127];
    v = v > 0.f ? v : NEG_POST * v;
    out[i] = v + xs[i];
}

// ---------------------------------------------------------------------------
// Host launcher
// ---------------------------------------------------------------------------
extern "C" void kernel_launch(void* const* d_in, const int* in_sizes, int n_in,
                              void* d_out, int out_size, void* d_ws, size_t ws_size,
                              hipStream_t stream)
{
    const float* x    = (const float*)d_in[0];
    const int*   ei   = (const int*)  d_in[1];
    const float* Wl   = (const float*)d_in[2];
    const float* bl   = (const float*)d_in[3];
    const float* Wr   = (const float*)d_in[4];
    const float* br   = (const float*)d_in[5];
    const float* att  = (const float*)d_in[6];
    const float* bias = (const float*)d_in[7];
    const float* Wsk  = (const float*)d_in[8];
    float* out = (float*)d_out;

    const int N  = in_sizes[0] / 128;
    const int E  = in_sizes[1] / 2;
    const int EN = E + N;
    const int nx = N * 128;
    const int nh = N * 4;

    // workspace carve-out (256B aligned slices)
    char* p = (char*)d_ws;
    auto carve = [&](size_t bytes) {
        char* r = p;
        p += (bytes + 255) & ~(size_t)255;
        return r;
    };
    __hip_bfloat16* xbf   = (__hip_bfloat16*)carve((size_t)N * 128 * 2);
    __hip_bfloat16* wcat  = (__hip_bfloat16*)carve((size_t)384 * 128 * 2);
    float* xl    = (float*)carve((size_t)N * 128 * 4);
    float* xr    = (float*)carve((size_t)N * 128 * 4);
    float* xs    = (float*)carve((size_t)N * 128 * 4);
    float* ebuf  = (float*)carve((size_t)EN * 4 * 4);
    float* emax  = (float*)carve((size_t)N * 4 * 4);
    float* denom = (float*)carve((size_t)N * 4 * 4);

    const int srcOff = 0, dstOff = E;

    gat_prep_kernel<<<(nx + 255) / 256, 256, 0, stream>>>(
        x, Wl, Wr, Wsk, xbf, wcat, emax, denom, out, nx, nh);

    gat_gemm_wmma_kernel<<<N / 16, 256, 0, stream>>>(
        (const __bf16*)xbf, (const __bf16*)wcat, bl, br, xl, xr, xs);

    const int nt = EN * 4;
    gat_edge_score_kernel<<<(nt + 255) / 256, 256, 0, stream>>>(
        ei + srcOff, ei + dstOff, xl, xr, att, ebuf, emax, E, EN);

    gat_edge_exp_kernel<<<(nt + 255) / 256, 256, 0, stream>>>(
        ei + dstOff, ebuf, emax, denom, E, EN);

    gat_edge_agg_kernel<<<(nt + 255) / 256, 256, 0, stream>>>(
        ei + srcOff, ei + dstOff, xl, ebuf, denom, out, E, EN);

    gat_final_kernel<<<(nx + 255) / 256, 256, 0, stream>>>(out, bias, xs, nx);
}
